// Encoder_9706626089319
// MI455X (gfx1250) — matrix operand order repaired and checked by reading, not compiled, not run on hardware
//
#include <hip/hip_runtime.h>
#include <hip/hip_bf16.h>

// ---------------------------------------------------------------------------
// Encoder block for MI455X (gfx1250, wave32, WMMA).
//   q,k,v = x@W{q,k,v}^T + b ; sims = q.k (no scale) ; softmax ; ctx = p.v ;
//   out = ctx@Wo^T + bo
// v_wmma_f32_16x16x32_f16 everywhere, LDS-tiled + double-buffered staging via
// CDNA5 global_load_async_to_lds_b128, flash-style streaming softmax.
// ---------------------------------------------------------------------------

typedef __attribute__((ext_vector_type(16))) _Float16 v16h;
typedef __attribute__((ext_vector_type(8)))  _Float16 v8h;
typedef __attribute__((ext_vector_type(8)))  float    v8f;

#define NBATCH 4
#define LSEQ   2048
#define HDIM   1024
#define NHEADS 16
#define HSZ    64
#define MTOT   (NBATCH * LSEQ)   // 8192 flattened rows

union Frag16 { v16h v; v8h h[2]; };

__device__ __forceinline__ v8h ldg8h(const _Float16* p) {
    return *reinterpret_cast<const v8h*>(p);
}

// --- CDNA5 async global->LDS copy (ISA §15.18.3, op 98; ASYNCcnt-tracked) --
__device__ __forceinline__ void cp16(const _Float16* g, _Float16* l) {
    asm volatile("global_load_async_to_lds_b128 %0, %1, off"
                 :
                 : "v"((unsigned int)(size_t)l),        // LDS byte address
                   "v"((unsigned long long)(size_t)g)   // 64-bit global VA
                 : "memory");
}
__device__ __forceinline__ void stage_commit() {
    asm volatile("s_wait_asynccnt 0x0" ::: "memory");
    __syncthreads();
}

// reductions across the 16-lane half-wave that holds one C-row (wave32)
__device__ __forceinline__ float redmax16(float v) {
#pragma unroll
    for (int m = 1; m < 16; m <<= 1) v = fmaxf(v, __shfl_xor(v, m, 32));
    return v;
}
__device__ __forceinline__ float redsum16(float v) {
#pragma unroll
    for (int m = 1; m < 16; m <<= 1) v += __shfl_xor(v, m, 32);
    return v;
}

// ---------------------------------------------------------------------------
// fp32 -> f16 elementwise convert (x and weights)
// ---------------------------------------------------------------------------
__global__ void __launch_bounds__(256)
cvt_f32_to_f16(const float* __restrict__ s, _Float16* __restrict__ d, int n) {
    for (int i = blockIdx.x * blockDim.x + threadIdx.x; i < n;
         i += gridDim.x * blockDim.x)
        d[i] = (_Float16)s[i];
}

// ---------------------------------------------------------------------------
// GEMM: out[m,n] = sum_k A[m,k] * W[n,k] + bias[n];  A: MxK, W: NxK, K=N=1024.
// Block = 8 waves = 128(M) x 64(N) tile; K-step 32; LDS double-buffered.
// Wave w owns rows [w*16, w*16+16); all waves share the B tile.
// TRANSPOSE=1 stores out[(nb*HDIM + n) * LSEQ + tok]  (for V).
// ---------------------------------------------------------------------------
template <typename OUTT, int TRANSPOSE>
__global__ void __launch_bounds__(256)
gemm_wmma(const _Float16* __restrict__ A, const _Float16* __restrict__ W,
          const float* __restrict__ bias, OUTT* __restrict__ out) {
    constexpr int K = HDIM, Ndim = HDIM;
    constexpr int BM = 128, BN = 64, BK = 32;
    constexpr int AST = BK + 8;                    // padded stride: conflict-free
    constexpr int BST = BK + 8;
    __shared__ __align__(16) _Float16 As[2][BM * AST];   // 2 x 10 KB
    __shared__ __align__(16) _Float16 Bs[2][BN * BST];   // 2 x  5 KB

    const int tid  = threadIdx.x;
    const int lane = tid & 31;
    const int wv   = tid >> 5;
    const int nblk = Ndim / BN;                    // 16
    const int bm0  = (int)(blockIdx.x / nblk) * BM;
    const int n0   = (int)(blockIdx.x % nblk) * BN;
    const int m    = lane & 15;
    const int hh   = lane >> 4;

    // staging roles: A tile = 512 16B chunks (2/thread), B tile = 256 (1/thread)
    const int ar = tid >> 2, ac = (tid & 3) * 8;   // A rows ar, ar+64
    const int br = tid >> 2, bc = (tid & 3) * 8;

    auto stage = [&](int buf, int k0) {
        cp16(A + (size_t)(bm0 + ar) * K + k0 + ac,      &As[buf][ar * AST + ac]);
        cp16(A + (size_t)(bm0 + ar + 64) * K + k0 + ac, &As[buf][(ar + 64) * AST + ac]);
        cp16(W + (size_t)(n0 + br) * K + k0 + bc,       &Bs[buf][br * BST + bc]);
    };

    v8f c[4];
#pragma unroll
    for (int t = 0; t < 4; ++t) {
        float bv = bias[n0 + t * 16 + m];          // N = lane%16, all rows
#pragma unroll
        for (int r = 0; r < 8; ++r) c[t][r] = bv;
    }

    stage(0, 0);
    stage_commit();
    int cur = 0;
    for (int k0 = 0; k0 < K; k0 += BK) {
        if (k0 + BK < K) stage(cur ^ 1, k0 + BK);

        Frag16 a;                                  // A 16x32: K = 16h + j
        const _Float16* arow = &As[cur][(wv * 16 + m) * AST];
        a.h[0] = *reinterpret_cast<const v8h*>(arow + 16 * hh);
        a.h[1] = *reinterpret_cast<const v8h*>(arow + 16 * hh + 8);
#pragma unroll
        for (int t = 0; t < 4; ++t) {
            const _Float16* brow = &Bs[cur][(t * 16 + m) * BST + 16 * hh];
            Frag16 b;                              // B 32x16: K = 16h + 2v + p
            b.h[0] = *reinterpret_cast<const v8h*>(brow);
            b.h[1] = *reinterpret_cast<const v8h*>(brow + 8);
            c[t] = __builtin_amdgcn_wmma_f32_16x16x32_f16(
                false, a.v, false, b.v, (short)0, c[t], false, false);
        }
        stage_commit();                            // publish next, retire cur
        cur ^= 1;
    }

#pragma unroll
    for (int t = 0; t < 4; ++t) {
#pragma unroll
        for (int r = 0; r < 8; ++r) {
            int row = bm0 + wv * 16 + r + 8 * hh;  // C layout: M = r + 8*(lane/16)
            int col = n0 + t * 16 + m;
            if (TRANSPOSE) {
                int nb = row / LSEQ, tok = row % LSEQ;
                out[((size_t)(nb * HDIM + col)) * LSEQ + tok] = (OUTT)c[t][r];
            } else {
                out[(size_t)row * Ndim + col] = (OUTT)c[t][r];
            }
        }
    }
}

// ---------------------------------------------------------------------------
// Flash attention. Block = 8 waves = 8 consecutive 16-row q-tiles of ONE
// (batch, head); the 32-token K-tile (32x64) and V-tile (64x32) are staged
// to LDS once per block and shared by all waves (8x load-traffic cut).
// V pre-transposed: Vt[(nb*HDIM + hs) * LSEQ + tok].
// ---------------------------------------------------------------------------
__global__ void __launch_bounds__(256)
attn_wmma(const _Float16* __restrict__ Qh, const _Float16* __restrict__ Kh,
          const _Float16* __restrict__ Vt, _Float16* __restrict__ Ctx) {
    constexpr int KST = HSZ + 8;                    // 72 halfs / row, 32 rows
    constexpr int VST = 32 + 8;                     // 40 halfs / row, 64 rows
    __shared__ __align__(16) _Float16 Ks[2][32 * KST];
    __shared__ __align__(16) _Float16 Vs[2][64 * VST];
    __shared__ __align__(16) _Float16 plds[8][16 * 32];  // per-wave P patch

    const int tid  = threadIdx.x;
    const int lane = tid & 31;
    const int wv   = tid >> 5;
    const int QB   = (LSEQ / 16) / 8;               // 16 blocks per (nb,h)
    const int b    = blockIdx.x;
    const int nb   = b / (NHEADS * QB);
    const int h    = (b / QB) % NHEADS;
    const int qt   = (b % QB) * 8 + wv;
    const int m    = lane & 15;
    const int hh   = lane >> 4;

    const _Float16* kbase = Kh + ((size_t)nb * LSEQ) * HDIM + h * HSZ;
    const _Float16* vbase = Vt + ((size_t)(nb * HDIM + h * HSZ)) * LSEQ;

    // staging roles: K tile 32x8 chunks, V tile 64x4 chunks; 1 chunk each
    const int kr = tid >> 3, kc = (tid & 7) * 8;
    const int vr = tid >> 2, vc = (tid & 3) * 8;
    auto stage = [&](int buf, int j0) {
        cp16(kbase + (size_t)(j0 + kr) * HDIM + kc, &Ks[buf][kr * KST + kc]);
        cp16(vbase + (size_t)vr * LSEQ + j0 + vc,   &Vs[buf][vr * VST + vc]);
    };

    // Q tile as two A fragments (k-dims 0..31 / 32..63), resident in VGPRs
    const _Float16* qrow =
        Qh + ((size_t)(nb * LSEQ + qt * 16 + m)) * HDIM + h * HSZ;
    Frag16 qa[2];
#pragma unroll
    for (int kk = 0; kk < 2; ++kk) {
        qa[kk].h[0] = ldg8h(qrow + kk * 32 + 16 * hh);
        qa[kk].h[1] = ldg8h(qrow + kk * 32 + 16 * hh + 8);
    }

    v8f acc[4];
#pragma unroll
    for (int t = 0; t < 4; ++t)
#pragma unroll
        for (int r = 0; r < 8; ++r) acc[t][r] = 0.0f;

    float mstat[8], lstat[8];
#pragma unroll
    for (int r = 0; r < 8; ++r) { mstat[r] = -3.0e38f; lstat[r] = 0.0f; }

    stage(0, 0);
    stage_commit();
    int cur = 0;
    for (int j0 = 0; j0 < LSEQ; j0 += 32) {
        if (j0 + 32 < LSEQ) stage(cur ^ 1, j0 + 32);

        // ---- S = Q . K^T : 2 token sub-tiles x 2 WMMA (K=64) --------------
        v8f s[2];
#pragma unroll
        for (int tt = 0; tt < 2; ++tt) {
            v8f z;
#pragma unroll
            for (int r = 0; r < 8; ++r) z[r] = 0.0f;
#pragma unroll
            for (int kk = 0; kk < 2; ++kk) {
                const _Float16* krow =
                    &Ks[cur][(tt * 16 + m) * KST + kk * 32 + 16 * hh];
                Frag16 bfr;                 // B[k][n] = K[tok=j0+tt*16+n][k]
                bfr.h[0] = *reinterpret_cast<const v8h*>(krow);
                bfr.h[1] = *reinterpret_cast<const v8h*>(krow + 8);
                z = __builtin_amdgcn_wmma_f32_16x16x32_f16(
                    false, qa[kk].v, false, bfr.v, (short)0, z, false, false);
            }
            s[tt] = z;
        }

        // ---- streaming softmax over these 32 columns ----------------------
        float scale[8];
#pragma unroll
        for (int r = 0; r < 8; ++r) {       // row M = r + 8*(lane/16)
            float rm = redmax16(fmaxf(s[0][r], s[1][r]));
            float mn = fmaxf(mstat[r], rm);
            scale[r] = __expf(mstat[r] - mn);
            float p0 = __expf(s[0][r] - mn);
            float p1 = __expf(s[1][r] - mn);
            s[0][r] = p0;
            s[1][r] = p1;
            lstat[r] = lstat[r] * scale[r] + redsum16(p0 + p1);
            mstat[r] = mn;
        }

        // ---- repack P: C-layout -> per-wave LDS patch -> A fragment -------
#pragma unroll
        for (int r = 0; r < 8; ++r) {
            plds[wv][(r + 8 * hh) * 32 + m]      = (_Float16)s[0][r];
            plds[wv][(r + 8 * hh) * 32 + 16 + m] = (_Float16)s[1][r];
        }
        asm volatile("s_wait_dscnt 0x0" ::: "memory");  // intra-wave LDS RAW
        Frag16 pa;
        pa.h[0] = *reinterpret_cast<const v8h*>(&plds[wv][m * 32 + 16 * hh]);
        pa.h[1] = *reinterpret_cast<const v8h*>(&plds[wv][m * 32 + 16 * hh + 8]);

        // ---- acc = acc*scale + P.V ----------------------------------------
#pragma unroll
        for (int t = 0; t < 4; ++t) {
#pragma unroll
            for (int r = 0; r < 8; ++r) acc[t][r] *= scale[r];
            const _Float16* vrow = &Vs[cur][(t * 16 + m) * VST + 16 * hh];
            Frag16 bfr;                     // B[k][n] = V[tok=j0+k][hs=t*16+n]
            bfr.h[0] = *reinterpret_cast<const v8h*>(vrow);
            bfr.h[1] = *reinterpret_cast<const v8h*>(vrow + 8);
            acc[t] = __builtin_amdgcn_wmma_f32_16x16x32_f16(
                false, pa.v, false, bfr.v, (short)0, acc[t], false, false);
        }

        stage_commit();
        cur ^= 1;
    }

    // ---- normalize & store ctx (f16, row-major (n*L) x H) -----------------
    float inv[8];
#pragma unroll
    for (int r = 0; r < 8; ++r) inv[r] = 1.0f / lstat[r];
    _Float16* crow = Ctx + ((size_t)(nb * LSEQ + qt * 16)) * HDIM + h * HSZ;
#pragma unroll
    for (int t = 0; t < 4; ++t)
#pragma unroll
        for (int r = 0; r < 8; ++r)
            crow[(size_t)(r + 8 * hh) * HDIM + t * 16 + m] =
                (_Float16)(acc[t][r] * inv[r]);
}

// ---------------------------------------------------------------------------
extern "C" void kernel_launch(void* const* d_in, const int* in_sizes, int n_in,
                              void* d_out, int out_size, void* d_ws,
                              size_t ws_size, hipStream_t stream) {
    const float* x  = (const float*)d_in[0];
    const float* Wq = (const float*)d_in[1];
    const float* bq = (const float*)d_in[2];
    const float* Wk = (const float*)d_in[3];
    const float* bk = (const float*)d_in[4];
    const float* Wv = (const float*)d_in[5];
    const float* bv = (const float*)d_in[6];
    const float* Wo = (const float*)d_in[7];
    const float* bo = (const float*)d_in[8];
    float* out = (float*)d_out;

    // workspace carve-up (f16 elements); total ~88 MB
    _Float16* w = (_Float16*)d_ws;
    size_t o = 0;
    _Float16* xh  = w + o; o += (size_t)MTOT * HDIM;
    _Float16* Wqh = w + o; o += (size_t)HDIM * HDIM;
    _Float16* Wkh = w + o; o += (size_t)HDIM * HDIM;
    _Float16* Wvh = w + o; o += (size_t)HDIM * HDIM;
    _Float16* Woh = w + o; o += (size_t)HDIM * HDIM;
    _Float16* Qh  = w + o; o += (size_t)MTOT * HDIM;
    _Float16* Kh  = w + o; o += (size_t)MTOT * HDIM;
    _Float16* Vt  = w + o; o += (size_t)MTOT * HDIM;   // transposed V
    _Float16* Ctx = w + o; o += (size_t)MTOT * HDIM;

    cvt_f32_to_f16<<<2048, 256, 0, stream>>>(x, xh, MTOT * HDIM);
    cvt_f32_to_f16<<<512, 256, 0, stream>>>(Wq, Wqh, HDIM * HDIM);
    cvt_f32_to_f16<<<512, 256, 0, stream>>>(Wk, Wkh, HDIM * HDIM);
    cvt_f32_to_f16<<<512, 256, 0, stream>>>(Wv, Wvh, HDIM * HDIM);
    cvt_f32_to_f16<<<512, 256, 0, stream>>>(Wo, Woh, HDIM * HDIM);

    // (MTOT/128) * (HDIM/64) = 64 * 16 = 1024 blocks of 8 waves
    const int gemm_blocks = (MTOT / 128) * (HDIM / 64);
    gemm_wmma<_Float16, 0><<<gemm_blocks, 256, 0, stream>>>(xh, Wqh, bq, Qh);
    gemm_wmma<_Float16, 0><<<gemm_blocks, 256, 0, stream>>>(xh, Wkh, bk, Kh);
    gemm_wmma<_Float16, 1><<<gemm_blocks, 256, 0, stream>>>(xh, Wvh, bv, Vt);

    // 4 * 16 * 16 = 1024 blocks; each block = 8 q-tiles of one (nb, head)
    attn_wmma<<<1024, 256, 0, stream>>>(Qh, Kh, Vt, Ctx);

    gemm_wmma<float, 0><<<gemm_blocks, 256, 0, stream>>>(Ctx, Woh, bo, out);
}
